// TransformerEncoder_55946243997883
// MI455X (gfx1250) — matrix operand order repaired and checked by reading, not compiled, not run on hardware
//
#include <hip/hip_runtime.h>
#include <cmath>

typedef __attribute__((ext_vector_type(16))) _Float16 v16h;
typedef __attribute__((ext_vector_type(8)))  float    v8f;
typedef unsigned int __attribute__((ext_vector_type(4))) u32x4;
typedef int          __attribute__((ext_vector_type(8))) i32x8;
typedef int          __attribute__((ext_vector_type(4))) i32x4;

#define TPB 256
#define BK  32

// Model dims
#define BB   4
#define SS   512
#define EE   768
#define HH   12
#define DD   64
#define FFD  3072
#define LL   12
#define SQRT_E 27.712812921102035f   // sqrt(768)

#if defined(__AMDGCN__) && __has_builtin(__builtin_amdgcn_tensor_load_to_lds)
#define HAVE_TDM 1
#else
#define HAVE_TDM 0
#endif

__device__ __forceinline__ float gelu_exact(float x) {
  return 0.5f * x * (1.0f + erff(x * 0.70710678118654752f));
}

#if HAVE_TDM
// ---------------------------------------------------------------------------
// TDM 2D tile load: rows x BK halves from global (row stride ld halves) into
// LDS at byte offset lds_off, with 16B padding after each 64B row so the LDS
// row stride is 40 halves (= LSW below). D# per CDNA5 ISA.
// Wave-level op (EXEC ignored): call from one wave per workgroup.
// ---------------------------------------------------------------------------
__device__ __forceinline__ void tdm_load_2d(unsigned lds_off, const void* gaddr,
                                            int rows, int ld) {
  unsigned long long ga = (unsigned long long)(size_t)gaddr;
  u32x4 g0 = { 1u,                                   // count=1 (valid descriptor)
               lds_off,                               // lds_addr (bytes)
               (unsigned)ga,                          // global_addr[31:0]
               (unsigned)(ga >> 32) | 0x80000000u };  // global_addr[56:32] | type=2
  i32x8 g1 = { (int)((1u << 16)   // data_size = 2 bytes
                   | (1u << 20)   // pad_enable
                   | (3u << 22)   // pad_interval: every 64 bytes
                   | (3u << 25)), // pad_amount: 4 DWORDs = 16 bytes
               (int)0xFFFF0000u,                 // tensor_dim0 lo16 (huge, no OOB)
               (int)0xFFFF7FFFu,                 // dim0 hi / dim1 lo
               (int)((32u << 16) | 0x7FFFu),     // tile_dim0=32 | dim1 hi
               rows,                             // tile_dim1 (tile_dim2=0)
               ld,                               // tensor_dim0_stride[31:0]
               0, 0 };
  i32x4 gz = { 0, 0, 0, 0 };
#if __clang_major__ >= 23
  i32x8 gz8 = { 0, 0, 0, 0, 0, 0, 0, 0 };
  __builtin_amdgcn_tensor_load_to_lds(g0, g1, gz, gz, gz8, 0);
#else
  __builtin_amdgcn_tensor_load_to_lds(g0, g1, gz, gz, 0);
#endif
}
__device__ __forceinline__ unsigned lds_off_of(const void* p) {
  return (unsigned)(unsigned long long)(size_t)p;   // low 32 bits = LDS offset
}
#endif

// ---------------------------------------------------------------------------
// WMMA GEMM:  C[M,N] = act( A[M,K] @ B^T + bias )
//   A: f16 [M,K] row-major (lda); B: f16 [N,K] row-major (ldb)  -> both tiles
//   are contiguous along K, staged by TDM (double-buffered) or uint4 copies.
//   C: f32 or f16 (OUT16). 8 wave32s; wave grid WROWS x WCOLS, FM x FN WMMAs.
// All dims are exact multiples of the tiles for this model (no bounds checks,
// EXEC stays all-ones as WMMA requires).
// ---------------------------------------------------------------------------
template<int WROWS, int WCOLS, int FM, int FN, bool HASBIAS, bool ACT_GELU, bool OUT16>
__global__ __launch_bounds__(TPB)
void gemm_wmma_kernel(const _Float16* __restrict__ A, const _Float16* __restrict__ Bm,
                      const float* __restrict__ bias, void* __restrict__ Cv,
                      int K, int lda, int ldb, int ldc,
                      long long sA, long long sB, long long sC)
{
  constexpr int MT  = WROWS * FM * 16;
  constexpr int NT  = WCOLS * FN * 16;
  constexpr int LSW = BK + 8;                 // LDS row stride (halves) = 80B

  __shared__ _Float16 lA[2][MT * LSW];
  __shared__ _Float16 lB[2][NT * LSW];

  const int tid  = threadIdx.x;
  const int lane = tid & 31;
  const int w    = tid >> 5;
  const int wr   = w / WCOLS;
  const int wc   = w % WCOLS;
  const int ln16 = lane & 15;
  const int hi   = lane >> 4;

  const int m0 = blockIdx.y * MT;
  const int n0 = blockIdx.x * NT;
  const int z  = blockIdx.z;

  const _Float16* Ag = A  + (long long)z * sA + (long long)m0 * lda;
  const _Float16* Bg = Bm + (long long)z * sB + (long long)n0 * ldb;

  v8f acc[FM][FN] = {};
  const int nsteps = K / BK;

#if HAVE_TDM
  if (w == 0) {                                // prologue: stage tiles for s=0
    tdm_load_2d(lds_off_of(&lA[0][0]), Ag, MT, lda);
    tdm_load_2d(lds_off_of(&lB[0][0]), Bg, NT, ldb);
  }
#endif

  for (int s = 0; s < nsteps; ++s) {
    const int cur = s & 1;
#if HAVE_TDM
    if (w == 0) {
      if (s + 1 < nsteps) {                    // issue next tile, retire current
        tdm_load_2d(lds_off_of(&lA[cur ^ 1][0]), Ag + (long long)(s + 1) * BK, MT, lda);
        tdm_load_2d(lds_off_of(&lB[cur ^ 1][0]), Bg + (long long)(s + 1) * BK, NT, ldb);
        __builtin_amdgcn_s_wait_tensorcnt(2);
      } else {
        __builtin_amdgcn_s_wait_tensorcnt(0);
      }
    }
    __syncthreads();
#else
    {
      const int k0 = s * BK;
      #pragma unroll
      for (int i = 0; i < (MT * 4) / TPB; ++i) {       // 16B chunks, 4 per row
        int c = tid + i * TPB;
        int m = c >> 2, q = c & 3;
        *(uint4*)&lA[cur][m * LSW + q * 8] =
            *(const uint4*)&Ag[(long long)m * lda + k0 + q * 8];
      }
      #pragma unroll
      for (int i = 0; i < (NT * 4) / TPB; ++i) {
        int c = tid + i * TPB;
        int n = c >> 2, q = c & 3;
        *(uint4*)&lB[cur][n * LSW + q * 8] =
            *(const uint4*)&Bg[(long long)n * ldb + k0 + q * 8];
      }
      __syncthreads();
    }
#endif

    // ---- fragments: k = 16*hi + element (contiguous), same order for A and B ----
    v16h af[FM], bf[FN];
    #pragma unroll
    for (int fm = 0; fm < FM; ++fm) {
      const _Float16* ap = &lA[cur][(wr * FM * 16 + fm * 16 + ln16) * LSW];
      #pragma unroll
      for (int g = 0; g < 8; ++g) {
        int kb = (hi << 4) + (g << 1);  // A 16x32 map
        af[fm][2 * g]     = ap[kb];
        af[fm][2 * g + 1] = ap[kb + 1];
      }
    }
    #pragma unroll
    for (int fn = 0; fn < FN; ++fn) {
      const _Float16* bp = &lB[cur][(wc * FN * 16 + fn * 16 + ln16) * LSW + (hi << 4)];
      #pragma unroll
      for (int h = 0; h < 16; ++h) bf[fn][h] = bp[h];           // K contiguous
    }

    #pragma unroll
    for (int fm = 0; fm < FM; ++fm)
      #pragma unroll
      for (int fn = 0; fn < FN; ++fn)
        acc[fm][fn] = __builtin_amdgcn_wmma_f32_16x16x32_f16(
            false, af[fm], false, bf[fn], (short)0, acc[fm][fn], false, false);

    __syncthreads();                           // protect buffer reuse
  }

  // ---- epilogue: C VGPR r -> M = r + hi*8, lane -> N ----
  #pragma unroll
  for (int fm = 0; fm < FM; ++fm) {
    int mrow = m0 + wr * FM * 16 + fm * 16 + hi * 8;
    #pragma unroll
    for (int fn = 0; fn < FN; ++fn) {
      int col = n0 + wc * FN * 16 + fn * 16 + ln16;
      float bv = HASBIAS ? bias[col] : 0.0f;
      #pragma unroll
      for (int r = 0; r < 8; ++r) {
        float v = acc[fm][fn][r] + bv;
        if (ACT_GELU) v = gelu_exact(v);
        long long off = (long long)z * sC + (long long)(mrow + r) * ldc + col;
        if (OUT16) ((_Float16*)Cv)[off] = (_Float16)v;
        else       ((float*)Cv)[off]    = v;
      }
    }
  }
}

// ---------------------------------------------------------------------------
// LayerNorm over E=768, one block per row; optional residual adds; f16 or f32 out
// ---------------------------------------------------------------------------
template<int NRES, bool OUT16>
__global__ __launch_bounds__(TPB)
void ln_kernel(const float* __restrict__ in, const float* __restrict__ gw,
               const float* __restrict__ bw, const float* __restrict__ r1,
               const float* __restrict__ r2, void* __restrict__ outv)
{
  const long long row = blockIdx.x;
  const int tid = threadIdx.x;
  const float* x = in + row * EE;

  float v0 = x[tid], v1 = x[tid + 256], v2 = x[tid + 512];
  __shared__ float rs[TPB], rq[TPB];
  rs[tid] = v0 + v1 + v2;
  rq[tid] = v0 * v0 + v1 * v1 + v2 * v2;
  __syncthreads();
  #pragma unroll
  for (int off = 128; off > 0; off >>= 1) {
    if (tid < off) { rs[tid] += rs[tid + off]; rq[tid] += rq[tid + off]; }
    __syncthreads();
  }
  const float mean = rs[0] * (1.0f / EE);
  const float var  = rq[0] * (1.0f / EE) - mean * mean;
  const float inv  = rsqrtf(var + 1e-5f);

  #pragma unroll
  for (int i = 0; i < 3; ++i) {
    int e = tid + i * 256;
    float val = (x[e] - mean) * inv * gw[e] + bw[e];
    if (NRES >= 1) val += r1[row * EE + e];
    if (NRES >= 2) val += r2[row * EE + e];
    if (OUT16) ((_Float16*)outv)[row * EE + e] = (_Float16)val;
    else       ((float*)outv)[row * EE + e]    = val;
  }
}

// ---------------------------------------------------------------------------
// softmax over last dim (512), then /sqrt(E) (post-softmax, faithful).
// Reads f32 scores, writes f16 probabilities. One wave per row, 8 rows/block.
// ---------------------------------------------------------------------------
__global__ __launch_bounds__(TPB)
void softmax_scale_kernel(const float* __restrict__ att, _Float16* __restrict__ out)
{
  const int tid  = threadIdx.x;
  const int lane = tid & 31;
  const int wv   = tid >> 5;
  const long long row = (long long)blockIdx.x * 8 + wv;
  const float* p  = att + row * SS;
  _Float16*   ph  = out + row * SS;

  float v[16];
  float mx = -1e30f;
  #pragma unroll
  for (int i = 0; i < 16; ++i) { v[i] = p[lane + i * 32]; mx = fmaxf(mx, v[i]); }
  #pragma unroll
  for (int m = 16; m >= 1; m >>= 1) mx = fmaxf(mx, __shfl_xor(mx, m, 32));
  float sum = 0.0f;
  #pragma unroll
  for (int i = 0; i < 16; ++i) { v[i] = __expf(v[i] - mx); sum += v[i]; }
  #pragma unroll
  for (int m = 16; m >= 1; m >>= 1) sum += __shfl_xor(sum, m, 32);
  const float scale = 1.0f / (sum * SQRT_E);
  #pragma unroll
  for (int i = 0; i < 16; ++i) ph[lane + i * 32] = (_Float16)(v[i] * scale);
}

// ---------------------------------------------------------------------------
// Weight convert + transpose: in f32 [K][N] -> out f16 [N][K]
// ---------------------------------------------------------------------------
__global__ __launch_bounds__(TPB)
void wcvt_t_kernel(const float* __restrict__ in, _Float16* __restrict__ out,
                   int N, int K)
{
  long long idx = (long long)blockIdx.x * TPB + threadIdx.x;   // over K*N
  int n = (int)(idx % N);
  long long k = idx / N;
  out[(long long)n * K + k] = (_Float16)in[idx];
}

// qkv [B,S,3E] (col = h*192 + d*3 + j) -> q,k [B,H,S,D] f16, v [B,H,D,S] f16
__global__ __launch_bounds__(TPB)
void qkv_split_kernel(const float* __restrict__ qkv, _Float16* __restrict__ q,
                      _Float16* __restrict__ k, _Float16* __restrict__ v)
{
  long long idx = (long long)blockIdx.x * TPB + threadIdx.x;   // over B*H*S*D
  int d = (int)(idx & 63);
  int s = (int)((idx >> 6) & 511);
  int h = (int)((idx >> 15) % HH);
  long long b = idx / ((long long)DD * SS * HH);
  long long in = (b * SS + s) * (3 * EE) + h * (DD * 3) + d * 3;
  q[idx] = (_Float16)qkv[in];
  k[idx] = (_Float16)qkv[in + 1];
  v[((b * HH + h) * DD + d) * SS + s] = (_Float16)qkv[in + 2];  // transposed
}

// oh f32 [B,H,S,D] -> o f16 [B,S,E]
__global__ __launch_bounds__(TPB)
void merge_heads_kernel(const float* __restrict__ oh, _Float16* __restrict__ o)
{
  long long idx = (long long)blockIdx.x * TPB + threadIdx.x;   // over B*S*E
  int e = (int)(idx % EE);
  long long bs = idx / EE;
  int h = e >> 6, d = e & 63;
  long long b = bs >> 9;
  int s = (int)(bs & 511);
  o[idx] = (_Float16)oh[(((b * HH + h) * SS) + s) * DD + d];
}

// ---------------------------------------------------------------------------
extern "C" void kernel_launch(void* const* d_in, const int* in_sizes, int n_in,
                              void* d_out, int out_size, void* d_ws, size_t ws_size,
                              hipStream_t stream)
{
  const float* X    = (const float*)d_in[0];
  const float* Wqkv = (const float*)d_in[1];
  const float* bqkv = (const float*)d_in[2];
  const float* Wp   = (const float*)d_in[3];
  const float* bp   = (const float*)d_in[4];
  const float* W1   = (const float*)d_in[5];
  const float* b1   = (const float*)d_in[6];
  const float* W2   = (const float*)d_in[7];
  const float* b2   = (const float*)d_in[8];
  const float* ln1g = (const float*)d_in[9];
  const float* ln1b = (const float*)d_in[10];
  const float* ln2g = (const float*)d_in[11];
  const float* ln2b = (const float*)d_in[12];
  const float* ln3g = (const float*)d_in[13];
  const float* ln3b = (const float*)d_in[14];
  const float* ln4g = (const float*)d_in[15];
  const float* ln4b = (const float*)d_in[16];

  const long long NBSE = (long long)BB * SS * EE;        // 1.57M (== B*H*S*D)
  const long long NQKV = (long long)BB * SS * 3 * EE;    // 4.72M
  const long long NFF  = (long long)BB * SS * FFD;       // 6.29M
  const long long NATT = (long long)BB * HH * SS * SS;   // 12.58M

  char* p = (char*)d_ws;
  auto allocf = [&](long long n) { float* r = (float*)p; p += n * 4; return r; };
  auto alloch = [&](long long n) { _Float16* r = (_Float16*)p; p += n * 2; return r; };

  // f32 regions
  float* xp0  = allocf(NBSE);
  float* xp1  = allocf(NBSE);
  float* r1b  = allocf(NBSE);
  float* big1 = allocf(NQKV);   // qkv out -> attn proj out -> f2
  float* big2 = allocf(NATT);   // scores -> oh (att@V out, first 1.57M)
  // f16 regions
  _Float16* wqkv16 = alloch((long long)EE * 3 * EE);   // [2304][768]
  _Float16* wp16   = alloch((long long)EE * EE);       // [768][768]
  _Float16* w116   = alloch((long long)EE * FFD);      // [3072][768]
  _Float16* w216   = alloch((long long)FFD * EE);      // [768][3072]
  _Float16* bh16   = alloch(NBSE);                     // LN1 out, later LN3 out
  _Float16* q16    = alloch(NBSE);
  _Float16* k16    = alloch(NBSE);
  _Float16* v16    = alloch(NBSE);                     // [B,H,D,S]
  _Float16* o16    = alloch(NBSE);
  _Float16* att16  = alloch(NATT);
  _Float16* f116   = alloch(NFF);

  if (ws_size < (size_t)(p - (char*)d_ws)) return;

  const int M = BB * SS;            // 2048
  const int SMROWS = BB * HH * SS;  // 24576 softmax rows

  const float* cur = X;
  for (int l = 0; l < LL; ++l) {
    const float* lWqkv = Wqkv + (long long)l * EE * 3 * EE;
    const float* lbqkv = bqkv + (long long)l * 3 * EE;
    const float* lWp   = Wp   + (long long)l * EE * EE;
    const float* lbp   = bp   + (long long)l * EE;
    const float* lW1   = W1   + (long long)l * EE * FFD;
    const float* lb1   = b1   + (long long)l * FFD;
    const float* lW2   = W2   + (long long)l * FFD * EE;
    const float* lb2   = b2   + (long long)l * EE;
    float* nxt = (l == LL - 1) ? (float*)d_out : ((l & 1) ? xp1 : xp0);

    // 0) weights -> f16, transposed to [N][K]
    wcvt_t_kernel<<<(int)((EE * 3 * EE) / TPB), TPB, 0, stream>>>(lWqkv, wqkv16, 3 * EE, EE);
    wcvt_t_kernel<<<(int)((EE * EE) / TPB),     TPB, 0, stream>>>(lWp,   wp16,   EE,     EE);
    wcvt_t_kernel<<<(int)((EE * FFD) / TPB),    TPB, 0, stream>>>(lW1,   w116,   FFD,    EE);
    wcvt_t_kernel<<<(int)((FFD * EE) / TPB),    TPB, 0, stream>>>(lW2,   w216,   EE,     FFD);

    // 1) h = LN1(x) -> f16
    ln_kernel<0, true><<<M, TPB, 0, stream>>>(cur, ln1g + l * EE, ln1b + l * EE,
                                              nullptr, nullptr, bh16);
    // 2) qkv = h @ Wqkv^T + bqkv -> f32 big1      [2048 x 2304]
    gemm_wmma_kernel<2,4,4,2,true,false,false>
        <<<dim3((3 * EE) / 128, M / 128, 1), TPB, 0, stream>>>(
        bh16, wqkv16, lbqkv, big1, EE, EE, EE, 3 * EE, 0, 0, 0);
    // 3) split -> q16,k16 [B,H,S,D], v16 [B,H,D,S]
    qkv_split_kernel<<<(int)(NBSE / TPB), TPB, 0, stream>>>(big1, q16, k16, v16);
    // 4) energy = Q @ K^T -> f32 big2 (batch 48)
    gemm_wmma_kernel<2,4,4,2,false,false,false>
        <<<dim3(SS / 128, SS / 128, BB * HH), TPB, 0, stream>>>(
        q16, k16, nullptr, big2, DD, DD, DD, SS,
        (long long)SS * DD, (long long)SS * DD, (long long)SS * SS);
    // 5) softmax + /sqrt(E): f32 scores -> f16 att16
    softmax_scale_kernel<<<SMROWS / 8, TPB, 0, stream>>>(big2, att16);
    // 6) oh = att @ V -> f32 into big2 (scores dead) (batch 48, N=64)
    gemm_wmma_kernel<4,2,2,2,false,false,false>
        <<<dim3(DD / 64, SS / 128, BB * HH), TPB, 0, stream>>>(
        att16, v16, nullptr, big2, SS, SS, SS, DD,
        (long long)SS * SS, (long long)SS * DD, (long long)SS * DD);
    // 7) merge heads -> o16 [2048 x 768] f16
    merge_heads_kernel<<<(int)(NBSE / TPB), TPB, 0, stream>>>(big2, o16);
    // 8) attn_out = o @ Wp^T + bp -> f32 big1
    gemm_wmma_kernel<2,4,4,2,true,false,false>
        <<<dim3(EE / 128, M / 128, 1), TPB, 0, stream>>>(
        o16, wp16, lbp, big1, EE, EE, EE, EE, 0, 0, 0);
    // 9) r1 = LN2(attn_out) + x -> f32
    ln_kernel<1, false><<<M, TPB, 0, stream>>>(big1, ln2g + l * EE, ln2b + l * EE,
                                               cur, nullptr, r1b);
    // 10) f = LN3(r1) -> f16
    ln_kernel<0, true><<<M, TPB, 0, stream>>>(r1b, ln3g + l * EE, ln3b + l * EE,
                                              nullptr, nullptr, bh16);
    // 11) f1 = gelu(f @ W1^T + b1) -> f16 f116   [2048 x 3072]
    gemm_wmma_kernel<2,4,4,2,true,true,true>
        <<<dim3(FFD / 128, M / 128, 1), TPB, 0, stream>>>(
        bh16, w116, lb1, f116, EE, EE, EE, FFD, 0, 0, 0);
    // 12) f2 = f1 @ W2^T + b2 -> f32 big1        [2048 x 768]
    gemm_wmma_kernel<2,4,4,2,true,false,false>
        <<<dim3(EE / 128, M / 128, 1), TPB, 0, stream>>>(
        f116, w216, lb2, big1, FFD, FFD, FFD, EE, 0, 0, 0);
    // 13) x_next = LN4(f2) + r1 + x
    ln_kernel<2, false><<<M, TPB, 0, stream>>>(big1, ln4g + l * EE, ln4b + l * EE,
                                               r1b, cur, nxt);
    cur = nxt;
  }
}
